// FourierMultiHeadSelfAttention_26963804684329
// MI455X (gfx1250) — matrix operand order repaired and checked by reading, not compiled, not run on hardware
//
#include <hip/hip_runtime.h>
#include <hip/hip_bf16.h>
#include <math.h>

// ---------------------------------------------------------------------------
// FourierMultiHeadSelfAttention  (B=8, L=4096, E=512, H=8, D=64, M=40)
//
// Only the 40 lowest rfft bins survive, and |conj(q)*k| = |q|*|k| makes the
// attention scores rank-1.  So the whole op is:
//   fwd  : X[c,0:40] = x[c,:] @ DFT_basis      (bf16 WMMA, K=4096, split-K=8)
//   attn : rank-1 softmax over D=64 + weighted v-sum (tiny VALU work)
//   inv  : out[l,c]  = inv_basis[l,:] @ Y[c,:]^T     (bf16 WMMA, K=96)
// Memory floor: 64MB x-read + 64MB out-write ~ 5.6us @ 23.3 TB/s; 5.4 GFLOP
// of bf16 WMMA is negligible vs that, so both GEMMs stream HBM exactly once.
// Split-K partials are summed in a separate kernel (deterministic, no float
// atomics), giving 2048 fwd waves instead of 256 to hide load latency.
// All six A tiles are preloaded into distinct registers per K-step so the
// compiler clauses the 12 b128 loads and issues 6 back-to-back WMMAs.
// ---------------------------------------------------------------------------

typedef __bf16 bf16_t;
typedef __attribute__((ext_vector_type(16))) __bf16 v16bf;
typedef __attribute__((ext_vector_type(8)))  float  v8f;

#define LSEQ   4096          // sequence length
#define EDIM   512           // embed
#define NB     8             // batch
#define NH     8             // heads
#define HD     64            // head dim
#define NMODE  40            // kept modes
#define NC     96            // padded mode columns (2*40 -> 96, mult of 32)
#define NCH    4096          // total channels = B*E
#define KSPLIT 8             // split-K slices for the forward GEMM
#define KCHUNK (LSEQ / KSPLIT)

// ------------------------------ basis tables -------------------------------
// WfT [NC][LSEQ]  : row 2k = cos(2pi k l/L), row 2k+1 = -sin(2pi k l/L)
// Winv[LSEQ][NC]  : col 2k = ck*cos, col 2k+1 = -ck*sin, ck = (k?2:1)/L
__global__ void build_basis(bf16_t* __restrict__ WfT, bf16_t* __restrict__ Winv) {
    int idx = blockIdx.x * blockDim.x + threadIdx.x;
    if (idx >= LSEQ * NC) return;
    int l = idx / NC, j = idx % NC;
    int k = j >> 1;
    float fwd = 0.f, inv = 0.f;
    if (j < 2 * NMODE) {
        int ph = (k * l) & (LSEQ - 1);                    // exact mod-L reduction
        float ang = (float)ph * (6.28318530717958647692f / (float)LSEQ);
        float cv = cosf(ang), sv = sinf(ang);
        float scale = (k == 0) ? (1.f / (float)LSEQ) : (2.f / (float)LSEQ);
        fwd = (j & 1) ? -sv : cv;
        inv = (j & 1) ? (-sv * scale) : (cv * scale);
    }
    WfT[(size_t)j * LSEQ + l] = (bf16_t)fwd;
    Winv[(size_t)l * NC + j]  = (bf16_t)inv;
}

// ------------------------------ forward DFT --------------------------------
// XmP[ks][j][c] = sum_{l in slice ks} WfT[j][l] * x[b,l,e] ;  c = b*E + e.
// One wave per (16-channel tile, K-slice). A (basis, K-contiguous) straight
// from L2 as uint4 (all 6 tiles preloaded -> one load clause, one wait);
// x tile transposed through LDS so the B pack is two ds_load_b128 per lane.
__global__ __launch_bounds__(32) void fwd_dft(const float* __restrict__ x,
                                              const bf16_t* __restrict__ WfT,
                                              float* __restrict__ XmP) {
    __shared__ bf16_t xs[16][40];          // [e-local][k-local], 80B rows (16B-aligned)
    const int c0 = blockIdx.x * 16;        // 256 channel tiles
    const int ks = blockIdx.y;             // K-slice
    const int b  = c0 / EDIM;
    const int e0 = c0 % EDIM;
    const int t  = threadIdx.x;
    const int lm = t & 15;                 // m (A) / n (B,C) within half-wave
    const int hi = t >> 4;

    v8f acc[6];
#pragma unroll
    for (int i = 0; i < 6; ++i) acc[i] = (v8f){0.f,0.f,0.f,0.f,0.f,0.f,0.f,0.f};

    const int lbeg = ks * KCHUNK;
    for (int l0 = lbeg; l0 < lbeg + KCHUNK; l0 += 32) {
        // stage x tile: lane t loads row l0+t, 16 floats, cvt -> bf16, LDS transposed
        const float4* xr = (const float4*)(x + ((size_t)b * LSEQ + (l0 + t)) * EDIM + e0);
        // prefetch next K-step's row (clamped in-bounds; lowers to global_prefetch_b8)
        int lpf = l0 + 32 < lbeg + KCHUNK ? l0 + 32 : l0;
        __builtin_prefetch(x + ((size_t)b * LSEQ + (lpf + t)) * EDIM + e0, 0, 0);
        float4 f0 = xr[0], f1 = xr[1], f2 = xr[2], f3 = xr[3];
        __syncthreads();
        xs[ 0][t] = (bf16_t)f0.x; xs[ 1][t] = (bf16_t)f0.y;
        xs[ 2][t] = (bf16_t)f0.z; xs[ 3][t] = (bf16_t)f0.w;
        xs[ 4][t] = (bf16_t)f1.x; xs[ 5][t] = (bf16_t)f1.y;
        xs[ 6][t] = (bf16_t)f1.z; xs[ 7][t] = (bf16_t)f1.w;
        xs[ 8][t] = (bf16_t)f2.x; xs[ 9][t] = (bf16_t)f2.y;
        xs[10][t] = (bf16_t)f2.z; xs[11][t] = (bf16_t)f2.w;
        xs[12][t] = (bf16_t)f3.x; xs[13][t] = (bf16_t)f3.y;
        xs[14][t] = (bf16_t)f3.z; xs[15][t] = (bf16_t)f3.w;
        __syncthreads();

        // pack B (32x16, K=l): lane holds K = hi*16+e for column lm -> 2x ds_load_b128
        v16bf bm;
        const bf16_t* bp = &xs[lm][hi * 16];
        ((uint4*)&bm)[0] = *(const uint4*)(bp);
        ((uint4*)&bm)[1] = *(const uint4*)(bp + 8);

        // preload ALL 6 basis A tiles (16x32, K-contiguous) -> single load clause
        v16bf am[6];
#pragma unroll
        for (int jt = 0; jt < 6; ++jt) {
            const bf16_t* ar = WfT + (size_t)(jt * 16 + lm) * LSEQ + l0 + hi * 16;
            ((uint4*)&am[jt])[0] = *(const uint4*)(ar);        // K = hi*16 + 0..7
            ((uint4*)&am[jt])[1] = *(const uint4*)(ar + 8);    // K = hi*16 + 8..15
        }
        // 6 back-to-back WMMAs sharing the B operand
#pragma unroll
        for (int jt = 0; jt < 6; ++jt)
            acc[jt] = __builtin_amdgcn_wmma_f32_16x16x32_bf16(
                false, am[jt], false, bm, (short)0, acc[jt], false, false);
    }
    // D layout: element r -> row hi*8+r (= mode col j), col lm (= channel)
    float* outp = XmP + (size_t)ks * NC * NCH;
#pragma unroll
    for (int jt = 0; jt < 6; ++jt)
#pragma unroll
        for (int r = 0; r < 8; ++r)
            outp[(size_t)(jt * 16 + hi * 8 + r) * NCH + c0 + lm] = acc[jt][r];
}

// -------------------------- split-K reduction ------------------------------
__global__ void reduce_k(const float* __restrict__ XmP, float* __restrict__ XmT) {
    int idx = blockIdx.x * blockDim.x + threadIdx.x;     // over NC*NCH
    if (idx >= NC * NCH) return;
    float s = 0.f;
#pragma unroll
    for (int ks = 0; ks < KSPLIT; ++ks) s += XmP[(size_t)ks * NC * NCH + idx];
    XmT[idx] = s;
}

// ------------------------------ attention ----------------------------------
// Per (b,h,m): scores[d,e] = |q_d|*|k_e| (rank-1), softmax over e, @ v, * Wo.
__global__ __launch_bounds__(64) void attn_k(const float* __restrict__ XmT,
                                             const float* __restrict__ Wq,
                                             const float* __restrict__ Wk,
                                             const float* __restrict__ Wv,
                                             const float* __restrict__ Wo,
                                             bf16_t* __restrict__ Y) {
    const int m = blockIdx.x, h = blockIdx.y, b = blockIdx.z;
    const int d = threadIdx.x;                        // 0..63
    const int c = b * EDIM + h * HD + d;
    const float xr = XmT[(size_t)(2 * m)     * NCH + c];
    const float xi = XmT[(size_t)(2 * m + 1) * NCH + c];
    const size_t wi = ((size_t)(h * HD + d) * NMODE + m) * 2;   // complex interleaved

    const float qr = xr * Wq[wi] - xi * Wq[wi + 1];
    const float qi = xr * Wq[wi + 1] + xi * Wq[wi];
    const float kr = xr * Wk[wi] - xi * Wk[wi + 1];
    const float ki = xr * Wk[wi + 1] + xi * Wk[wi];
    const float vr = xr * Wv[wi] - xi * Wv[wi + 1];
    const float vi = xr * Wv[wi + 1] + xi * Wv[wi];
    const float amag = sqrtf(qr * qr + qi * qi);      // |q_d|
    const float kmag = sqrtf(kr * kr + ki * ki);      // |k_d|

    __shared__ float skm[HD], svr[HD], svi[HD];
    skm[d] = kmag; svr[d] = vr; svi[d] = vi;
    __syncthreads();

    float kmax = 0.f;
#pragma unroll 8
    for (int e = 0; e < HD; ++e) kmax = fmaxf(kmax, skm[e]);
    const float mx = amag * kmax;                     // a_d >= 0, k_e >= 0

    float s = 0.f, sr = 0.f, si = 0.f;
#pragma unroll 4
    for (int e = 0; e < HD; ++e) {
        float w = __expf(amag * skm[e] - mx);
        s += w; sr += w * svr[e]; si += w * svi[e];
    }
    const float inv = 1.f / s;
    const float xqr = sr * inv, xqi = si * inv;
    const float orr = xqr * Wo[wi] - xqi * Wo[wi + 1];
    const float oi  = xqr * Wo[wi + 1] + xqi * Wo[wi];
    Y[(size_t)c * NC + 2 * m]     = (bf16_t)orr;      // pad cols 80..95 pre-zeroed
    Y[(size_t)c * NC + 2 * m + 1] = (bf16_t)oi;
}

// ------------------------------ inverse DFT --------------------------------
// out[b,l,e] = sum_j Winv[l,j] * Y[c,j]  (K=96, 3 WMMA steps, store-dominated)
// 4 waves per block, each wave owns one 16x16 output tile (its own l-tile).
__global__ __launch_bounds__(128) void inv_dft(const bf16_t* __restrict__ Winv,
                                               const bf16_t* __restrict__ Y,
                                               float* __restrict__ out) {
    const int wid = threadIdx.x >> 5;       // wave id 0..3
    const int t   = threadIdx.x & 31;
    const int e0  = blockIdx.x * 16;        // 32 e-tiles
    const int l0  = (blockIdx.y * 4 + wid) * 16;   // 256 l-tiles
    const int b   = blockIdx.z;
    const int lm  = t & 15;
    const int hi  = t >> 4;
    const int c0  = b * EDIM + e0;

    v8f acc = (v8f){0.f,0.f,0.f,0.f,0.f,0.f,0.f,0.f};
#pragma unroll
    for (int k0 = 0; k0 < NC; k0 += 32) {
        // A: row l0+lm of Winv, K-contiguous
        const bf16_t* ar = Winv + (size_t)(l0 + lm) * NC + k0 + hi * 16;
        v16bf am;
        ((uint4*)&am)[0] = *(const uint4*)(ar);
        ((uint4*)&am)[1] = *(const uint4*)(ar + 8);
        // B: column = channel c0+lm, K-contiguous in Y row
        const bf16_t* br = Y + (size_t)(c0 + lm) * NC + k0 + hi * 16;
        v16bf bm;
        ((uint4*)&bm)[0] = *(const uint4*)(br);
        ((uint4*)&bm)[1] = *(const uint4*)(br + 8);
        acc = __builtin_amdgcn_wmma_f32_16x16x32_bf16(
            false, am, false, bm, (short)0, acc, false, false);
    }
#pragma unroll
    for (int r = 0; r < 8; ++r)
        out[((size_t)b * LSEQ + l0 + hi * 8 + r) * EDIM + e0 + lm] = acc[r];
}

// ------------------------------ launcher -----------------------------------
extern "C" void kernel_launch(void* const* d_in, const int* in_sizes, int n_in,
                              void* d_out, int out_size, void* d_ws, size_t ws_size,
                              hipStream_t stream) {
    const float* x  = (const float*)d_in[0];
    const float* Wq = (const float*)d_in[1];   // complex64 -> interleaved floats
    const float* Wk = (const float*)d_in[2];
    const float* Wv = (const float*)d_in[3];
    const float* Wo = (const float*)d_in[4];

    char* ws = (char*)d_ws;
    const size_t szWfT  = (size_t)NC * LSEQ * sizeof(bf16_t);        // 768 KB
    const size_t szWinv = (size_t)LSEQ * NC * sizeof(bf16_t);        // 768 KB
    const size_t szXmT  = (size_t)NC * NCH * sizeof(float);          // 1.5 MB
    const size_t szY    = (size_t)NCH * NC * sizeof(bf16_t);         // 768 KB
    bf16_t* WfT  = (bf16_t*)(ws);
    bf16_t* Winv = (bf16_t*)(ws + szWfT);
    float*  XmT  = (float*)(ws + szWfT + szWinv);
    bf16_t* Y    = (bf16_t*)(ws + szWfT + szWinv + szXmT);
    float*  XmP  = (float*)(ws + szWfT + szWinv + szXmT + szY);      // 12.6 MB

    // zero Y so the 16 padded K-columns are 0 for the inverse GEMM
    hipMemsetAsync(Y, 0, szY, stream);

    build_basis<<<(LSEQ * NC + 255) / 256, 256, 0, stream>>>(WfT, Winv);
    fwd_dft<<<dim3(NCH / 16, KSPLIT), 32, 0, stream>>>(x, WfT, XmP);
    reduce_k<<<(NC * NCH + 255) / 256, 256, 0, stream>>>(XmP, XmT);
    attn_k<<<dim3(NMODE, NH, NB), HD, 0, stream>>>(XmT, Wq, Wk, Wv, Wo, Y);
    inv_dft<<<dim3(EDIM / 16, LSEQ / 64, NB), 128, 0, stream>>>(Winv, Y, (float*)d_out);
}
